// SimpleALIFRNN_89756226552395
// MI455X (gfx1250) — matrix operand order repaired and checked by reading, not compiled, not run on hardware
//
#include <hip/hip_runtime.h>
#include <hip/hip_bf16.h>
#include <math.h>

// ---------------- problem constants ----------------
#define TT 250
#define BB 256
#define II 700
#define HH 1024
#define OO 20
#define KH (II + HH)      // 1724
#define KPAD 1728          // next multiple of 32

typedef __attribute__((ext_vector_type(16))) __bf16 v16bf;
typedef __attribute__((ext_vector_type(8)))  __bf16 v8bf;
typedef __attribute__((ext_vector_type(8)))  float  v8f;

// ---------------- WMMA helpers ----------------
__device__ inline v8f wmma_bf16(v16bf a, v16bf b, v8f c) {
  // D(f32 16x16) = A(bf16 16x32) * B(bf16 32x16) + C
  return __builtin_amdgcn_wmma_f32_16x16x32_bf16(
      /*neg_a=*/false, a, /*neg_b=*/false, b,
      /*c_mod=*/(short)0, c, /*reuse_a=*/false, /*reuse_b=*/false);
}

__device__ inline v16bf cat16(v8bf lo, v8bf hi) {
  v16bf f;
#pragma unroll
  for (int i = 0; i < 8; ++i) { f[i] = lo[i]; f[i + 8] = hi[i]; }
  return f;
}

// A-matrix fragment (16x32 bf16, row-major source, ld = KPAD).
// ISA layout: lanes 0-15 -> row M=lane, K = {kc..kc+7, kc+16..kc+23}
//             lanes 16-31 -> row M=lane-16, K = {kc+8..kc+15, kc+24..kc+31}
__device__ inline v16bf load_frag_a(const __bf16* __restrict__ base, int row0,
                                    int kc, int lane) {
  int r = lane & 15, hi = lane >> 4;
  const __bf16* p = base + (size_t)(row0 + r) * KPAD + kc + hi * 8;
  v8bf lo = *(const v8bf*)p;
  v8bf hv = *(const v8bf*)(p + 16);
  return cat16(lo, hv);
}

// B-matrix fragment (32x16 bf16). Source is W_h native layout [N][K], ld=KPAD,
// so column n of B == row n of W_h (no transpose needed).
// ISA layout: lanes 0-15 -> col N=lane, K = {kc..kc+7, kc+16..kc+23}
//             lanes 16-31 -> col N=lane-16, K = {kc+8..kc+15, kc+24..kc+31}
__device__ inline v16bf load_frag_b(const __bf16* __restrict__ base, int col0,
                                    int kc, int lane) {
  int r = lane & 15, hi = lane >> 4;
  const __bf16* p = base + (size_t)(col0 + r) * KPAD + kc + hi * 8;
  v8bf lo = *(const v8bf*)p;
  v8bf hv = *(const v8bf*)(p + 16);
  return cat16(lo, hv);
}

// ---------------- one-time prep kernels ----------------
__global__ void convert_w_kernel(const float* __restrict__ W_h,
                                 __bf16* __restrict__ wb) {
  int idx = blockIdx.x * blockDim.x + threadIdx.x;
  if (idx >= HH * KPAD) return;
  int n = idx / KPAD, k = idx - n * KPAD;
  float v = (k < KH) ? W_h[(size_t)n * KH + k] : 0.0f;
  wb[idx] = (__bf16)v;
}

__global__ void decay_kernel(const float* __restrict__ tau_mem,
                             const float* __restrict__ tau_adp,
                             const float* __restrict__ tau_out,
                             float* __restrict__ alpha, float* __restrict__ rho,
                             float* __restrict__ alpha_o) {
  int tid = threadIdx.x;
  if (tid < HH) {
    alpha[tid] = expf(-1.0f / tau_mem[tid]);
    rho[tid]   = expf(-1.0f / tau_adp[tid]);
  }
  if (tid < OO) alpha_o[tid] = expf(-1.0f / tau_out[tid]);
}

__global__ void init_state_kernel(float* __restrict__ state, int n,
                                  unsigned int* __restrict__ spk) {
  int idx = blockIdx.x * blockDim.x + threadIdx.x;
  if (idx < n) state[idx] = 0.0f;
  if (idx == 0) *spk = 0u;
}

// ---------------- per-step kernels ----------------
// Build A = [bf16(x_t) | bf16(z_prev) | zero-pad] in [BB][KPAD].
__global__ void pack_a_kernel(const float* __restrict__ x_t,
                              const float* __restrict__ z,
                              __bf16* __restrict__ ab) {
  int idx = blockIdx.x * blockDim.x + threadIdx.x;  // grid sized exactly
  int b = idx / KPAD, k = idx - b * KPAD;
  float v;
  if (k < II)       v = x_t[(size_t)b * II + k];
  else if (k < KH)  v = z[(size_t)b * HH + (k - II)];
  else              v = 0.0f;
  ab[idx] = (__bf16)v;
}

// Fused recurrent GEMM (bf16 WMMA, fp32 accumulate) + ALIF state update.
// Grid: (BB/64, HH/64); block: 128 threads = 4 waves, each wave 32x32.
__global__ void __launch_bounds__(128)
alif_gemm_kernel(const __bf16* __restrict__ ab, const __bf16* __restrict__ wb,
                 const float* __restrict__ alpha, const float* __restrict__ rho,
                 float* __restrict__ z, float* __restrict__ u,
                 float* __restrict__ a, unsigned int* __restrict__ spk) {
  const int lane = threadIdx.x & 31;
  const int wave = threadIdx.x >> 5;
  const int rowBase = blockIdx.x * 64 + (wave & 1) * 32;   // batch rows
  const int colBase = blockIdx.y * 64 + (wave >> 1) * 32;  // hidden cols

  v8f zero8 = {0.f, 0.f, 0.f, 0.f, 0.f, 0.f, 0.f, 0.f};
  v8f acc[2][2] = {{zero8, zero8}, {zero8, zero8}};

#pragma unroll 2
  for (int kc = 0; kc < KPAD; kc += 32) {
    v16bf a0 = load_frag_a(ab, rowBase,      kc, lane);
    v16bf a1 = load_frag_a(ab, rowBase + 16, kc, lane);
    v16bf b0 = load_frag_b(wb, colBase,      kc, lane);
    v16bf b1 = load_frag_b(wb, colBase + 16, kc, lane);
    acc[0][0] = wmma_bf16(a0, b0, acc[0][0]);
    acc[0][1] = wmma_bf16(a0, b1, acc[0][1]);
    acc[1][0] = wmma_bf16(a1, b0, acc[1][0]);
    acc[1][1] = wmma_bf16(a1, b1, acc[1][1]);
  }

  // Epilogue: C layout -> lane holds column n = lane&15, rows m = (lane>>4)*8+v.
  unsigned int spikes = 0;
  const int n  = lane & 15;
  const int mB = (lane >> 4) * 8;
#pragma unroll
  for (int nt = 0; nt < 2; ++nt) {
    int h = colBase + nt * 16 + n;
    float al = alpha[h], rh = rho[h];
#pragma unroll
    for (int mt = 0; mt < 2; ++mt) {
#pragma unroll
      for (int v = 0; v < 8; ++v) {
        int b = rowBase + mt * 16 + mB + v;
        size_t idx = (size_t)b * HH + h;
        float cur = acc[mt][nt][v];
        float zo = z[idx];
        float an = rh * a[idx] + (1.0f - rh) * zo;       // adaptation
        float th = 0.01f + 1.8f * an;                    // adaptive threshold
        float un = al * u[idx] + (1.0f - al) * cur - zo * th;
        float zn = (un - th) > 0.0f ? 1.0f : 0.0f;       // spike
        z[idx] = zn; u[idx] = un; a[idx] = an;
        spikes += (unsigned int)zn;
      }
    }
  }
#pragma unroll
  for (int off = 16; off > 0; off >>= 1)
    spikes += __shfl_xor(spikes, off, 32);
  if (lane == 0) atomicAdd(spk, spikes);
}

// Leaky-integrator readout: one wave per output neuron, one block per batch row.
__global__ void readout_kernel(const float* __restrict__ z,
                               const float* __restrict__ W_o,
                               const float* __restrict__ alpha_o,
                               float* __restrict__ ou,
                               float* __restrict__ out_t) {
  const int b = blockIdx.x;
  const int o = threadIdx.x >> 5;
  const int lane = threadIdx.x & 31;
  const float* zr = z + (size_t)b * HH;
  const float* wr = W_o + (size_t)o * HH;
  float s = 0.0f;
#pragma unroll 4
  for (int h = lane; h < HH; h += 32) s += zr[h] * wr[h];
#pragma unroll
  for (int off = 16; off > 0; off >>= 1) s += __shfl_xor(s, off, 32);
  if (lane == 0) {
    float ao = alpha_o[o];
    float v = ao * ou[b * OO + o] + (1.0f - ao) * s;
    ou[b * OO + o] = v;
    out_t[b * OO + o] = v;
  }
}

__global__ void finalize_kernel(const unsigned int* __restrict__ spk,
                                float* __restrict__ nspk) {
  if (threadIdx.x == 0 && blockIdx.x == 0) *nspk = (float)(*spk);
}

// ---------------- host launcher ----------------
extern "C" void kernel_launch(void* const* d_in, const int* in_sizes, int n_in,
                              void* d_out, int out_size, void* d_ws,
                              size_t ws_size, hipStream_t stream) {
  const float* x       = (const float*)d_in[0];  // [T,B,I]
  const float* W_h     = (const float*)d_in[1];  // [H, I+H]
  const float* tau_mem = (const float*)d_in[2];  // [H]
  const float* tau_adp = (const float*)d_in[3];  // [H]
  const float* W_o     = (const float*)d_in[4];  // [O, H]
  const float* tau_out = (const float*)d_in[5];  // [O]

  // d_out layout (return-order concat): outputs, zT, uT, aT, ouT, num_spikes
  float* out      = (float*)d_out;
  float* outputs  = out;                          // T*B*O
  float* zs       = outputs + (size_t)TT * BB * OO;
  float* us       = zs + (size_t)BB * HH;
  float* as_      = us + (size_t)BB * HH;
  float* ous      = as_ + (size_t)BB * HH;
  float* nspk     = ous + (size_t)BB * OO;

  // workspace: ab | wb | alpha | rho | alpha_o | spike counter  (~4.5 MB)
  char* ws = (char*)d_ws;
  __bf16* ab = (__bf16*)ws;                                   // BB*KPAD
  __bf16* wb = (__bf16*)(ws + (size_t)BB * KPAD * 2);         // HH*KPAD
  float* alpha   = (float*)(ws + (size_t)BB * KPAD * 2 + (size_t)HH * KPAD * 2);
  float* rho     = alpha + HH;
  float* alpha_o = rho + HH;
  unsigned int* spk = (unsigned int*)(alpha_o + 32);

  // one-time prep (per launch, deterministic)
  convert_w_kernel<<<(HH * KPAD + 255) / 256, 256, 0, stream>>>(W_h, wb);
  decay_kernel<<<1, 1024, 0, stream>>>(tau_mem, tau_adp, tau_out, alpha, rho,
                                       alpha_o);
  const int nstate = 3 * BB * HH + BB * OO;
  init_state_kernel<<<(nstate + 255) / 256, 256, 0, stream>>>(zs, nstate, spk);

  dim3 ggrid(BB / 64, HH / 64);  // 4 x 16
  for (int t = 0; t < TT; ++t) {
    pack_a_kernel<<<(BB * KPAD) / 256, 256, 0, stream>>>(
        x + (size_t)t * BB * II, zs, ab);
    alif_gemm_kernel<<<ggrid, 128, 0, stream>>>(ab, wb, alpha, rho, zs, us,
                                                as_, spk);
    readout_kernel<<<BB, OO * 32, 0, stream>>>(zs, W_o, alpha_o, ous,
                                               outputs + (size_t)t * BB * OO);
  }
  finalize_kernel<<<1, 1, 0, stream>>>(spk, nspk);
}
